// EncodeProcessDecode_60696477827374
// MI455X (gfx1250) — matrix operand order repaired and checked by reading, not compiled, not run on hardware
//
#include <hip/hip_runtime.h>
#include <hip/hip_bf16.h>

// ---------------------------------------------------------------------------
// EncodeProcessDecode GNN for gfx1250 (MI455X).
// Heavy GEMMs (all MLP layers with 128-wide outputs) run on bf16 WMMA
// (v_wmma_f32_16x16x32_bf16, f32 accumulate). Graph gathers are fused into
// GEMM A-tile staging via "concat segment" descriptors, so the big concat
// inputs (up to 80000 x 1024) are never materialized. Each 256-thread block
// computes a 64x128 output stripe: 4 WMMAs per wave per K-step share one
// staged A tile and one register-held B fragment.
// ---------------------------------------------------------------------------

typedef __bf16 bf16_t;
typedef bf16_t bf16x16 __attribute__((ext_vector_type(16)));
typedef float  f32x8   __attribute__((ext_vector_type(8)));

#define N_EDGES  80000
#define N_NODES  10000
#define N_GRAPHS 16
#define LAT      128
#define MROWS    64            // rows per block
#define MTILES   (MROWS / 16)  // 16-row WMMA subtiles per block

struct Seg  { const float* src; const int* idx; int width; };
struct Segs { Seg s[8]; int n; };

// ---------------------------------------------------------------------------
// Weight transpose + fp32->bf16 convert: W [K][N] row-major -> Wt [N][Kpad]
// ---------------------------------------------------------------------------
__global__ __launch_bounds__(256)
void transpose_cvt(const float* __restrict__ W, bf16_t* __restrict__ Wt,
                   int K, int Kpad, int N) {
  int t = blockIdx.x * 256 + threadIdx.x;
  if (t >= N * Kpad) return;
  int n = t / Kpad, k = t % Kpad;
  Wt[t] = (k < K) ? (bf16_t)W[(size_t)k * N + n] : (bf16_t)0.0f;
}

// ---------------------------------------------------------------------------
// edge_batch[e] = batch[row[e]]
// ---------------------------------------------------------------------------
__global__ __launch_bounds__(256)
void compute_edge_batch(const int* __restrict__ row, const int* __restrict__ batch,
                        int* __restrict__ eb, int E) {
  int t = blockIdx.x * 256 + threadIdx.x;
  if (t < E) eb[t] = batch[row[t]];
}

// ---------------------------------------------------------------------------
// Generic WMMA GEMM layer: out = [LN](relu(concat(segs) @ Wt^T + bias))
//   A: M x Ktot (fp32, gathered per segment, converted to bf16 in LDS)
//   Wt: bf16 [128][Kpad] (pre-transposed weights), Kpad % 32 == 0
//   do_ln: apply per-row LayerNorm over the 128 features (gamma/beta)
// Block: 256 threads = 8 waves; block computes MROWS rows x 128 cols.
// Wave w owns columns [16w,16w+16); per K-step it issues MTILES wmmas that
// reuse one B fragment and one staged A tile.
// ---------------------------------------------------------------------------
__global__ __launch_bounds__(256)
void gemm_mlp_layer(Segs segs, int M, int Ktot, int Kpad,
                    const bf16_t* __restrict__ Wt, const float* __restrict__ bias,
                    const float* __restrict__ gamma, const float* __restrict__ beta,
                    float* __restrict__ out, int do_ln) {
  __shared__ __attribute__((aligned(16))) bf16_t Atile[MROWS * 32];
  __shared__ float Y[MROWS * 128];
  __shared__ float ln_mu[MROWS], ln_rs[MROWS];

  const int wave = threadIdx.x >> 5;
  const int lane = threadIdx.x & 31;
  const int rl   = lane & 15;       // row within A-frag / col within tile
  const int hi   = lane >> 4;       // K-half selector
  const int m0   = blockIdx.x * MROWS;
  const int col  = wave * 16 + rl;  // global output column (0..127)

  f32x8 acc[MTILES];
  #pragma unroll
  for (int mt = 0; mt < MTILES; ++mt) acc[mt] = (f32x8){};

  for (int kt = 0; kt < Kpad; kt += 32) {
    // ---- stage A tile (MROWS x 32 k) with gather + cvt to bf16 ----
    for (int i = threadIdx.x; i < MROWS * 32; i += 256) {
      int r = i >> 5, k = i & 31;
      int kg = kt + k;
      float v = 0.0f;
      int rowg = m0 + r;
      if (rowg < M && kg < Ktot) {
        int off = kg;
        #pragma unroll 1
        for (int s = 0; s < segs.n; ++s) {
          int w = segs.s[s].width;
          if (off < w) {
            const int* idx = segs.s[s].idx;
            int sr = idx ? idx[rowg] : rowg;
            v = segs.s[s].src[(size_t)sr * w + off];
            break;
          }
          off -= w;
        }
      }
      Atile[r * 32 + k] = (bf16_t)v;
    }
    __syncthreads();

    // ---- B fragment (32x16 bf16) straight from transposed weights ----
    union { bf16x16 v; uint4 q[2]; } bv;
    const bf16_t* wrow = Wt + (size_t)col * Kpad + kt + hi * 16;  // 32B aligned
    bv.q[0] = reinterpret_cast<const uint4*>(wrow)[0];
    bv.q[1] = reinterpret_cast<const uint4*>(wrow)[1];

    // ---- MTILES A fragments + WMMAs reusing the same B fragment ----
    #pragma unroll
    for (int mt = 0; mt < MTILES; ++mt) {
      union { bf16x16 v; unsigned u[8]; } av;
      const bf16_t* arow = &Atile[(mt * 16 + rl) * 32];
      #pragma unroll
      for (int j = 0; j < 8; ++j) {
        int k0 = hi * 16 + j * 2;  // even -> 4B aligned
        av.u[j] = *reinterpret_cast<const unsigned*>(&arow[k0]);
      }
      acc[mt] = __builtin_amdgcn_wmma_f32_16x16x32_bf16(
          /*neg_a=*/false, av.v, /*neg_b=*/false, bv.v,
          /*c_mod=*/(short)0, acc[mt], /*reuse_a=*/false, /*reuse_b=*/false);
    }
    __syncthreads();
  }

  // ---- epilogue: bias + relu (+ LayerNorm) ----
  #pragma unroll
  for (int mt = 0; mt < MTILES; ++mt) {
    #pragma unroll
    for (int v = 0; v < 8; ++v) {
      int rloc = mt * 16 + v + hi * 8;            // C/D layout: M = v (+8 hi lanes)
      float val = acc[mt][v] + bias[col];
      val = fmaxf(val, 0.0f);
      if (!do_ln) {
        int rowg = m0 + rloc;
        if (rowg < M) out[(size_t)rowg * 128 + col] = val;
      } else {
        Y[rloc * 128 + col] = val;
      }
    }
  }
  if (do_ln) {
    __syncthreads();
    if (threadIdx.x < MROWS) {
      const float* yr = &Y[threadIdx.x * 128];
      float mu = 0.0f;
      for (int c = 0; c < 128; ++c) mu += yr[c];
      mu *= (1.0f / 128.0f);
      float var = 0.0f;
      for (int c = 0; c < 128; ++c) { float d = yr[c] - mu; var += d * d; }
      var *= (1.0f / 128.0f);
      ln_mu[threadIdx.x] = mu;
      ln_rs[threadIdx.x] = rsqrtf(var + 1e-5f);
    }
    __syncthreads();
    for (int i = threadIdx.x; i < MROWS * 128; i += 256) {
      int rloc = i >> 7, c = i & 127;
      int rowg = m0 + rloc;
      if (rowg < M)
        out[(size_t)rowg * 128 + c] =
            (Y[i] - ln_mu[rloc]) * ln_rs[rloc] * gamma[c] + beta[c];
    }
  }
}

// ---------------------------------------------------------------------------
// Segment-sum scatter: dst[idx[m]][c] += src[m][c]   (global_atomic_add_f32)
// ---------------------------------------------------------------------------
__global__ __launch_bounds__(256)
void scatter_add(const float* __restrict__ src, const int* __restrict__ idx,
                 float* __restrict__ dst, int M, int C) {
  int t = blockIdx.x * 256 + threadIdx.x;
  if (t >= M * C) return;
  int m = t / C, c = t % C;
  atomicAdd(&dst[(size_t)idx[m] * C + c], src[t]);
}

// ---------------------------------------------------------------------------
// Small (M=16) two-layer MLP + LayerNorm for the global update.
// One block per graph; 128 threads (one per latent feature).
// ---------------------------------------------------------------------------
__global__ __launch_bounds__(128)
void global_mlp(Segs segs, int Ktot,
                const float* __restrict__ W1, const float* __restrict__ b1,
                const float* __restrict__ W2, const float* __restrict__ b2,
                const float* __restrict__ g, const float* __restrict__ be,
                float* __restrict__ out) {
  __shared__ float x[512];
  __shared__ float h[128];
  __shared__ float y[128];
  __shared__ float stats[2];
  int rowg = blockIdx.x;
  int tid = threadIdx.x;
  for (int k = tid; k < Ktot; k += 128) {
    int off = k; float v = 0.0f;
    for (int s = 0; s < segs.n; ++s) {
      int w = segs.s[s].width;
      if (off < w) {
        const int* idx = segs.s[s].idx;
        int sr = idx ? idx[rowg] : rowg;
        v = segs.s[s].src[(size_t)sr * w + off];
        break;
      }
      off -= w;
    }
    x[k] = v;
  }
  __syncthreads();
  float a = b1[tid];
  for (int k = 0; k < Ktot; ++k) a += x[k] * W1[(size_t)k * 128 + tid];
  h[tid] = fmaxf(a, 0.0f);
  __syncthreads();
  float a2 = b2[tid];
  for (int j = 0; j < 128; ++j) a2 += h[j] * W2[(size_t)j * 128 + tid];
  y[tid] = fmaxf(a2, 0.0f);
  __syncthreads();
  if (tid == 0) {
    float mu = 0.0f;
    for (int c = 0; c < 128; ++c) mu += y[c];
    mu *= (1.0f / 128.0f);
    float var = 0.0f;
    for (int c = 0; c < 128; ++c) { float d = y[c] - mu; var += d * d; }
    var *= (1.0f / 128.0f);
    stats[0] = mu;
    stats[1] = rsqrtf(var + 1e-5f);
  }
  __syncthreads();
  out[(size_t)rowg * 128 + tid] = (y[tid] - stats[0]) * stats[1] * g[tid] + be[tid];
}

// ---------------------------------------------------------------------------
// Output block (single Linear, tiny N) scalar kernels.
// ---------------------------------------------------------------------------
__global__ __launch_bounds__(256)
void out_edge_kernel(const float* __restrict__ nd, const float* __restrict__ ed,
                     const float* __restrict__ gd, const int* __restrict__ row,
                     const int* __restrict__ col, const int* __restrict__ eb,
                     const float* __restrict__ W, const float* __restrict__ b,
                     float* __restrict__ out, int E) {
  int t = blockIdx.x * 256 + threadIdx.x;
  if (t >= E * 3) return;
  int e = t / 3, o = t % 3;
  const float* nr = nd + (size_t)col[e] * 128;
  const float* ns = nd + (size_t)row[e] * 128;
  const float* ee = ed + (size_t)e * 128;
  const float* gg = gd + (size_t)eb[e] * 128;
  float acc = b[o];
  for (int k = 0; k < 128; ++k) acc += nr[k] * W[k * 3 + o];
  for (int k = 0; k < 128; ++k) acc += ns[k] * W[(128 + k) * 3 + o];
  for (int k = 0; k < 128; ++k) acc += ee[k] * W[(256 + k) * 3 + o];
  for (int k = 0; k < 128; ++k) acc += gg[k] * W[(384 + k) * 3 + o];
  out[t] = acc;
}

__global__ __launch_bounds__(256)
void out_node_kernel(const float* __restrict__ nd, const float* __restrict__ gd,
                     const int* __restrict__ batch,
                     const float* __restrict__ recv3, const float* __restrict__ send3,
                     const float* __restrict__ W, const float* __restrict__ b,
                     float* __restrict__ out, int N) {
  int t = blockIdx.x * 256 + threadIdx.x;
  if (t >= N * 3) return;
  int n = t / 3, o = t % 3;
  const float* nn = nd + (size_t)n * 128;
  const float* gg = gd + (size_t)batch[n] * 128;
  float acc = b[o];
  for (int k = 0; k < 128; ++k) acc += nn[k] * W[k * 3 + o];
  for (int k = 0; k < 128; ++k) acc += gg[k] * W[(128 + k) * 3 + o];
  for (int k = 0; k < 3; ++k)   acc += recv3[n * 3 + k] * W[(256 + k) * 3 + o];
  for (int k = 0; k < 3; ++k)   acc += send3[n * 3 + k] * W[(259 + k) * 3 + o];
  out[t] = acc;
}

__global__ __launch_bounds__(32)
void out_global_kernel(const float* __restrict__ n2g3, const float* __restrict__ e2g3,
                       const float* __restrict__ gd,
                       const float* __restrict__ W, const float* __restrict__ b,
                       float* __restrict__ out) {
  int t = threadIdx.x;
  if (t >= N_GRAPHS * 2) return;
  int gidx = t / 2, o = t % 2;
  float acc = b[o];
  for (int k = 0; k < 3; ++k)   acc += n2g3[gidx * 3 + k] * W[k * 2 + o];
  for (int k = 0; k < 3; ++k)   acc += e2g3[gidx * 3 + k] * W[(3 + k) * 2 + o];
  for (int k = 0; k < 128; ++k) acc += gd[(size_t)gidx * 128 + k] * W[(6 + k) * 2 + o];
  out[t] = acc;
}

// ---------------------------------------------------------------------------
// Host orchestration
// ---------------------------------------------------------------------------
struct MLPW { const float *W1, *b1, *W2, *b2, *g, *be; };

static inline Seg seg(const float* p, const int* i, int w) {
  Seg s; s.src = p; s.idx = i; s.width = w; return s;
}

extern "C" void kernel_launch(void* const* d_in, const int* in_sizes, int n_in,
                              void* d_out, int out_size, void* d_ws, size_t ws_size,
                              hipStream_t stream) {
  (void)in_sizes; (void)n_in; (void)out_size; (void)ws_size;

  const float* edge_attr   = (const float*)d_in[0];  // [E,4]
  const float* node_attr   = (const float*)d_in[1];  // [N,8]
  const float* global_attr = (const float*)d_in[2];  // [G,4]
  const int*   ei          = (const int*)d_in[3];    // [2,E]
  const int*   batch       = (const int*)d_in[4];    // [N]
  const int* row = ei;
  const int* col = ei + N_EDGES;

  int pi = 5;
  auto take = [&](MLPW& m) {
    m.W1 = (const float*)d_in[pi]; m.b1 = (const float*)d_in[pi + 1];
    m.W2 = (const float*)d_in[pi + 2]; m.b2 = (const float*)d_in[pi + 3];
    m.g  = (const float*)d_in[pi + 4]; m.be = (const float*)d_in[pi + 5];
    pi += 6;
  };
  MLPW enc_e, enc_n, enc_g, pro_e, pro_n, pro_g, dec_e, dec_n, dec_g;
  take(enc_e); take(enc_n); take(enc_g);
  take(pro_e); take(pro_n); take(pro_g);
  take(dec_e); take(dec_n); take(dec_g);
  const float* outW_e = (const float*)d_in[pi++]; const float* outb_e = (const float*)d_in[pi++];
  const float* outW_n = (const float*)d_in[pi++]; const float* outb_n = (const float*)d_in[pi++];
  const float* outW_g = (const float*)d_in[pi++]; const float* outb_g = (const float*)d_in[pi++];

  // ---- workspace bump allocator ----
  char* wp = (char*)d_ws;
  auto alloc = [&](size_t bytes) -> void* {
    void* p = (void*)wp;
    wp += (bytes + 255) & ~(size_t)255;
    return p;
  };
  const size_t EB = (size_t)N_EDGES * LAT * sizeof(float);
  const size_t NB = (size_t)N_NODES * LAT * sizeof(float);
  const size_t GB = (size_t)N_GRAPHS * LAT * sizeof(float);
  float* He   = (float*)alloc(EB);   // MLP hidden (reused, sized for edges)
  float* e0   = (float*)alloc(EB);
  float* eA   = (float*)alloc(EB);
  float* eB   = (float*)alloc(EB);
  float* ed   = (float*)alloc(EB);
  float* n0   = (float*)alloc(NB);
  float* nA   = (float*)alloc(NB);
  float* nB   = (float*)alloc(NB);
  float* nd   = (float*)alloc(NB);
  float* recv = (float*)alloc(NB);
  float* send = (float*)alloc(NB);
  float* g0   = (float*)alloc(GB);
  float* gA   = (float*)alloc(GB);
  float* gB   = (float*)alloc(GB);
  float* gd   = (float*)alloc(GB);
  float* n2g  = (float*)alloc(GB);
  float* e2g  = (float*)alloc(GB);
  float* recv3 = (float*)alloc((size_t)N_NODES * 3 * sizeof(float));
  float* send3 = (float*)alloc((size_t)N_NODES * 3 * sizeof(float));
  float* n2g3  = (float*)alloc((size_t)N_GRAPHS * 3 * sizeof(float));
  float* e2g3  = (float*)alloc((size_t)N_GRAPHS * 3 * sizeof(float));
  int*   eb    = (int*)alloc((size_t)N_EDGES * sizeof(int));

  auto kp = [](int k) { return (k + 31) & ~31; };
  auto prep = [&](const float* W, int K) -> bf16_t* {
    int Kpad = kp(K);
    bf16_t* t = (bf16_t*)alloc((size_t)LAT * Kpad * sizeof(bf16_t));
    int tot = LAT * Kpad;
    transpose_cvt<<<(tot + 255) / 256, 256, 0, stream>>>(W, t, K, Kpad, LAT);
    return t;
  };
  // weight prep (transpose + bf16) for the WMMA GEMMs
  bf16_t* enc_eW1t = prep(enc_e.W1, 24);    bf16_t* enc_eW2t = prep(enc_e.W2, 128);
  bf16_t* enc_nW1t = prep(enc_n.W1, 268);   bf16_t* enc_nW2t = prep(enc_n.W2, 128);
  bf16_t* pro_eW1t = prep(pro_e.W1, 1024);  bf16_t* pro_eW2t = prep(pro_e.W2, 128);
  bf16_t* pro_nW1t = prep(pro_n.W1, 768);   bf16_t* pro_nW2t = prep(pro_n.W2, 128);
  bf16_t* dec_eW1t = prep(dec_e.W1, 512);   bf16_t* dec_eW2t = prep(dec_e.W2, 128);
  bf16_t* dec_nW1t = prep(dec_n.W1, 512);   bf16_t* dec_nW2t = prep(dec_n.W2, 128);

  compute_edge_batch<<<(N_EDGES + 255) / 256, 256, 0, stream>>>(row, batch, eb, N_EDGES);

  auto run_mlp = [&](const Segs& s1, int K1, int M, bf16_t* W1t, const MLPW& m,
                     bf16_t* W2t, float* out) {
    gemm_mlp_layer<<<(M + MROWS - 1) / MROWS, 256, 0, stream>>>(
        s1, M, K1, kp(K1), W1t, m.b1, nullptr, nullptr, He, 0);
    Segs s2; s2.n = 1; s2.s[0] = seg(He, nullptr, LAT);
    gemm_mlp_layer<<<(M + MROWS - 1) / MROWS, 256, 0, stream>>>(
        s2, M, LAT, LAT, W2t, m.b2, m.g, m.be, out, 1);
  };
  auto agg_edges = [&](const float* e_new) {
    hipMemsetAsync(recv, 0, NB, stream);
    hipMemsetAsync(send, 0, NB, stream);
    int tot = N_EDGES * LAT;
    scatter_add<<<(tot + 255) / 256, 256, 0, stream>>>(e_new, col, recv, N_EDGES, LAT);
    scatter_add<<<(tot + 255) / 256, 256, 0, stream>>>(e_new, row, send, N_EDGES, LAT);
  };
  auto agg_glob = [&](const float* n_new, const float* e_new) {
    hipMemsetAsync(n2g, 0, GB, stream);
    hipMemsetAsync(e2g, 0, GB, stream);
    scatter_add<<<(N_NODES * LAT + 255) / 256, 256, 0, stream>>>(n_new, batch, n2g, N_NODES, LAT);
    scatter_add<<<(N_EDGES * LAT + 255) / 256, 256, 0, stream>>>(e_new, eb, e2g, N_EDGES, LAT);
  };

  // ================= encoder =================
  {
    Segs se; se.n = 4;
    se.s[0] = seg(node_attr, col, 8);
    se.s[1] = seg(node_attr, row, 8);
    se.s[2] = seg(edge_attr, nullptr, 4);
    se.s[3] = seg(global_attr, eb, 4);
    run_mlp(se, 24, N_EDGES, enc_eW1t, enc_e, enc_eW2t, e0);
    agg_edges(e0);
    Segs sn; sn.n = 4;
    sn.s[0] = seg(node_attr, nullptr, 8);
    sn.s[1] = seg(global_attr, batch, 4);
    sn.s[2] = seg(recv, nullptr, LAT);
    sn.s[3] = seg(send, nullptr, LAT);
    run_mlp(sn, 268, N_NODES, enc_nW1t, enc_n, enc_nW2t, n0);
    agg_glob(n0, e0);
    Segs sg; sg.n = 3;
    sg.s[0] = seg(n2g, nullptr, LAT);
    sg.s[1] = seg(e2g, nullptr, LAT);
    sg.s[2] = seg(global_attr, nullptr, 4);
    global_mlp<<<N_GRAPHS, 128, 0, stream>>>(sg, 260, enc_g.W1, enc_g.b1,
                                             enc_g.W2, enc_g.b2, enc_g.g, enc_g.be, g0);
  }

  // ================= processing steps =================
  float* e_cur = e0; float* n_cur = n0; float* g_cur = g0;
  for (int step = 0; step < 3; ++step) {
    float* e_nx = (e_cur == eA) ? eB : eA;
    float* n_nx = (n_cur == nA) ? nB : nA;
    float* g_nx = (g_cur == gA) ? gB : gA;

    // -------- processor --------
    {
      Segs pe; pe.n = 8;
      pe.s[0] = seg(n0, col, LAT);    pe.s[1] = seg(n_cur, col, LAT);
      pe.s[2] = seg(n0, row, LAT);    pe.s[3] = seg(n_cur, row, LAT);
      pe.s[4] = seg(e0, nullptr, LAT); pe.s[5] = seg(e_cur, nullptr, LAT);
      pe.s[6] = seg(g0, eb, LAT);     pe.s[7] = seg(g_cur, eb, LAT);
      run_mlp(pe, 1024, N_EDGES, pro_eW1t, pro_e, pro_eW2t, e_nx);
      agg_edges(e_nx);
      Segs pn; pn.n = 6;
      pn.s[0] = seg(n0, nullptr, LAT);  pn.s[1] = seg(n_cur, nullptr, LAT);
      pn.s[2] = seg(g0, batch, LAT);    pn.s[3] = seg(g_cur, batch, LAT);
      pn.s[4] = seg(recv, nullptr, LAT); pn.s[5] = seg(send, nullptr, LAT);
      run_mlp(pn, 768, N_NODES, pro_nW1t, pro_n, pro_nW2t, n_nx);
      agg_glob(n_nx, e_nx);
      Segs pg; pg.n = 4;
      pg.s[0] = seg(n2g, nullptr, LAT); pg.s[1] = seg(e2g, nullptr, LAT);
      pg.s[2] = seg(g0, nullptr, LAT);  pg.s[3] = seg(g_cur, nullptr, LAT);
      global_mlp<<<N_GRAPHS, 128, 0, stream>>>(pg, 512, pro_g.W1, pro_g.b1,
                                               pro_g.W2, pro_g.b2, pro_g.g, pro_g.be, g_nx);
      e_cur = e_nx; n_cur = n_nx; g_cur = g_nx;
    }

    // -------- decoder --------
    {
      Segs de; de.n = 4;
      de.s[0] = seg(n_cur, col, LAT);
      de.s[1] = seg(n_cur, row, LAT);
      de.s[2] = seg(e_cur, nullptr, LAT);
      de.s[3] = seg(g_cur, eb, LAT);
      run_mlp(de, 512, N_EDGES, dec_eW1t, dec_e, dec_eW2t, ed);
      agg_edges(ed);
      Segs dn; dn.n = 4;
      dn.s[0] = seg(n_cur, nullptr, LAT);
      dn.s[1] = seg(g_cur, batch, LAT);
      dn.s[2] = seg(recv, nullptr, LAT);
      dn.s[3] = seg(send, nullptr, LAT);
      run_mlp(dn, 512, N_NODES, dec_nW1t, dec_n, dec_nW2t, nd);
      agg_glob(nd, ed);
      Segs dg; dg.n = 3;
      dg.s[0] = seg(n2g, nullptr, LAT); dg.s[1] = seg(e2g, nullptr, LAT);
      dg.s[2] = seg(g_cur, nullptr, LAT);
      global_mlp<<<N_GRAPHS, 128, 0, stream>>>(dg, 384, dec_g.W1, dec_g.b1,
                                               dec_g.W2, dec_g.b2, dec_g.g, dec_g.be, gd);
    }

    // -------- output block (only the last step's result is returned) --------
    if (step == 2) {
      float* oute = (float*)d_out;                              // [E,3]
      float* outn = (float*)d_out + (size_t)N_EDGES * 3;        // [N,3]
      float* outg = (float*)d_out + (size_t)N_EDGES * 3 + (size_t)N_NODES * 3;  // [G,2]
      out_edge_kernel<<<(N_EDGES * 3 + 255) / 256, 256, 0, stream>>>(
          nd, ed, gd, row, col, eb, outW_e, outb_e, oute, N_EDGES);
      hipMemsetAsync(recv3, 0, (size_t)N_NODES * 3 * sizeof(float), stream);
      hipMemsetAsync(send3, 0, (size_t)N_NODES * 3 * sizeof(float), stream);
      scatter_add<<<(N_EDGES * 3 + 255) / 256, 256, 0, stream>>>(oute, col, recv3, N_EDGES, 3);
      scatter_add<<<(N_EDGES * 3 + 255) / 256, 256, 0, stream>>>(oute, row, send3, N_EDGES, 3);
      out_node_kernel<<<(N_NODES * 3 + 255) / 256, 256, 0, stream>>>(
          nd, gd, batch, recv3, send3, outW_n, outb_n, outn, N_NODES);
      hipMemsetAsync(n2g3, 0, (size_t)N_GRAPHS * 3 * sizeof(float), stream);
      hipMemsetAsync(e2g3, 0, (size_t)N_GRAPHS * 3 * sizeof(float), stream);
      scatter_add<<<(N_NODES * 3 + 255) / 256, 256, 0, stream>>>(outn, batch, n2g3, N_NODES, 3);
      scatter_add<<<(N_EDGES * 3 + 255) / 256, 256, 0, stream>>>(oute, eb, e2g3, N_EDGES, 3);
      out_global_kernel<<<1, 32, 0, stream>>>(n2g3, e2g3, gd, outW_g, outb_g, outg);
    }
  }
}
